// LSTM_61847529062500
// MI455X (gfx1250) — matrix operand order repaired and checked by reading, not compiled, not run on hardware
//
#include <hip/hip_runtime.h>
#include <hip/hip_bf16.h>
#include <math.h>

#define Hdim 1024
#define Bsz  1024
#define Ssz  128
#define Edim 6
#define Vdim 3
#define Cdim 10

typedef __attribute__((ext_vector_type(16))) __bf16 v16bf;
typedef __attribute__((ext_vector_type(8)))  __bf16 v8bf;
typedef __attribute__((ext_vector_type(8)))  float  v8f;
typedef __attribute__((ext_vector_type(4)))  float  v4f;

union V16U { v16bf v; v8bf h[2]; };
union V8FU { v8f v; v4f q[2]; };

// Fast device transcendentals on the v_exp_f32 path (saturate correctly).
__device__ __forceinline__ float sigmoid_f(float x) {
    return 1.0f / (1.0f + __expf(-x));
}
__device__ __forceinline__ float tanh_f(float x) {
    // tanh(x) = 2/(1+exp(-2x)) - 1
    return fmaf(2.0f, 1.0f / (1.0f + __expf(-2.0f * x)), -1.0f);
}

// ---------------------------------------------------------------------------
// Prologue: convert the 4 recurrent weight matrices f32 -> bf16 (gate-major).
// ---------------------------------------------------------------------------
__global__ __launch_bounds__(256)
void conv_weights(const float* __restrict__ wg, const float* __restrict__ wi,
                  const float* __restrict__ wf, const float* __restrict__ wo,
                  __bf16* __restrict__ out)
{
    const int idx = blockIdx.x * 256 + threadIdx.x;           // [0, H*H)
    out[0 * Hdim * Hdim + idx] = (__bf16)wg[idx];
    out[1 * Hdim * Hdim + idx] = (__bf16)wi[idx];
    out[2 * Hdim * Hdim + idx] = (__bf16)wf[idx];
    out[3 * Hdim * Hdim + idx] = (__bf16)wo[idx];
}

// ---------------------------------------------------------------------------
// Prologue: V=3, so the per-step input projection collapses to a table:
//   Xproj[g][tok][m] = b_g[m] + sum_e W_gx[m,e] * emb[tok,e]   (4 x 3 x 1024)
// ---------------------------------------------------------------------------
__global__ __launch_bounds__(256)
void xproj_build(const float* __restrict__ Wgx, const float* __restrict__ bg,
                 const float* __restrict__ Wix, const float* __restrict__ bi,
                 const float* __restrict__ Wfx, const float* __restrict__ bff,
                 const float* __restrict__ Wox, const float* __restrict__ bo,
                 const float* __restrict__ emb,
                 float* __restrict__ xproj)
{
    const int idx = blockIdx.x * 256 + threadIdx.x;           // [0, 4*3*H)
    const int g   = idx / (Vdim * Hdim);
    const int tok = (idx / Hdim) % Vdim;
    const int m   = idx % Hdim;
    const float* Wx[4] = { Wgx, Wix, Wfx, Wox };
    const float* bb[4] = { bg,  bi,  bff, bo  };
    float s = bb[g][m];
    #pragma unroll
    for (int e = 0; e < Edim; ++e)
        s = fmaf(Wx[g][m * Edim + e], emb[tok * Edim + e], s);
    xproj[idx] = s;
}

// ---------------------------------------------------------------------------
// Prologue: transpose h_init/c_init (H,B) -> batch-major (B,H); h as bf16.
// ---------------------------------------------------------------------------
__global__ __launch_bounds__(256)
void init_state(const float* __restrict__ h0, const float* __restrict__ c0,
                __bf16* __restrict__ hT, float* __restrict__ cT)
{
    const int idx = blockIdx.x * 256 + threadIdx.x;           // [0, B*H)
    const int b = idx / Hdim;
    const int h = idx % Hdim;
    hT[idx] = (__bf16)h0[h * Bsz + b];
    cT[idx] = c0[h * Bsz + b];
}

// ---------------------------------------------------------------------------
// One LSTM time step. grid = (B/64, H/64), block = 256 (8 wave32).
// Wave w: m-tile = (w&3)*16, n-tile = (w>>2)*32 (two 16-wide WMMA n tiles).
// hT layout is (B,H) bf16 so the WMMA B-operand is one contiguous 32B load.
// ---------------------------------------------------------------------------
__global__ __launch_bounds__(256)
void lstm_step(const __bf16* __restrict__ Wbf,   // 4*H*H bf16, gates g,i,f,o
               const __bf16* __restrict__ hT_in, // (B,H) bf16
               __bf16* __restrict__ hT_out,      // (B,H) bf16
               float*  __restrict__ cT,          // (B,H) f32 (in-place)
               const int*   __restrict__ x,      // (B,S) int32
               const float* __restrict__ xproj,  // (4,3,H) f32
               int t)
{
    const int tid  = threadIdx.x;
    const int wave = tid >> 5;
    const int lane = tid & 31;
    const int lm   = lane & 15;
    const bool hi  = lane >= 16;

    const int m0 = blockIdx.y * 64 + (wave & 3) * 16;
    const int n0 = blockIdx.x * 64 + (wave >> 2) * 32;

    // A-operand (16x32 bf16): lane<16 holds row m0+lm, K 0..15;
    // lane>=16 same row, K 16..31.
    const int arow  = m0 + lm;
    const int acol0 = hi ? 16 : 0;
    // C/D: element v of the f32x8 accumulator is row m0 + v + (hi?8:0), col lm.
    const int mbase = m0 + (hi ? 8 : 0);
    // B-operand (32x16 bf16): lane<16 -> col n, K 0..15; lane>=16 -> K 16..31.
    const int bk = hi ? 16 : 0;
    const int ncol[2] = { n0 + lm, n0 + 16 + lm };

    // ---- accumulator init from the precomputed projection table ----
    const int tok0 = x[ncol[0] * Ssz + t];
    const int tok1 = x[ncol[1] * Ssz + t];
    v8f acc[4][2];
    #pragma unroll
    for (int g = 0; g < 4; ++g) {
        const float* x0 = xproj + ((size_t)(g * Vdim + tok0)) * Hdim + mbase;
        const float* x1 = xproj + ((size_t)(g * Vdim + tok1)) * Hdim + mbase;
        V8FU a0, a1;
        a0.q[0] = *(const v4f*)(x0);
        a0.q[1] = *(const v4f*)(x0 + 4);
        a1.q[0] = *(const v4f*)(x1);
        a1.q[1] = *(const v4f*)(x1 + 4);
        acc[g][0] = a0.v;
        acc[g][1] = a1.v;
    }

    // ---- main GEMM: 4 gates, K = 1024, 8 WMMAs per k-step ----
    const __bf16* bptr0 = hT_in + (size_t)ncol[0] * Hdim + bk;
    const __bf16* bptr1 = hT_in + (size_t)ncol[1] * Hdim + bk;
    const __bf16* aptr  = Wbf + (size_t)arow * Hdim + acol0;

    #pragma unroll 4
    for (int k = 0; k < Hdim; k += 32) {
        const v16bf bm0 = *(const v16bf*)(bptr0 + k);
        const v16bf bm1 = *(const v16bf*)(bptr1 + k);
        #pragma unroll
        for (int g = 0; g < 4; ++g) {
            const __bf16* ap = aptr + (size_t)g * Hdim * Hdim + k;
            V16U a;
            a.h[0] = *(const v8bf*)(ap);
            a.h[1] = *(const v8bf*)(ap + 8);
            acc[g][0] = __builtin_amdgcn_wmma_f32_16x16x32_bf16(
                false, a.v, false, bm0, (short)0, acc[g][0], false, false);
            acc[g][1] = __builtin_amdgcn_wmma_f32_16x16x32_bf16(
                false, a.v, false, bm1, (short)0, acc[g][1], false, false);
        }
    }

    // ---- cell update: c2 = tanh(g)*sig(i) + c*sig(f); h2 = tanh(c2)*sig(o) ----
    #pragma unroll
    for (int nt = 0; nt < 2; ++nt) {
        float* cp = cT + (size_t)ncol[nt] * Hdim + mbase;
        float cv[8];
        *(v4f*)(cv + 0) = *(const v4f*)(cp + 0);
        *(v4f*)(cv + 4) = *(const v4f*)(cp + 4);
        v8bf hb;
        #pragma unroll
        for (int v = 0; v < 8; ++v) {
            const float gg = tanh_f(acc[0][nt][v]);
            const float ii = sigmoid_f(acc[1][nt][v]);
            const float ff = sigmoid_f(acc[2][nt][v]);
            const float oo = sigmoid_f(acc[3][nt][v]);
            const float c2 = gg * ii + cv[v] * ff;
            cv[v] = c2;
            hb[v] = (__bf16)(tanh_f(c2) * oo);
        }
        *(v4f*)(cp + 0) = *(const v4f*)(cv + 0);
        *(v4f*)(cp + 4) = *(const v4f*)(cv + 4);
        *(v8bf*)(hT_out + (size_t)ncol[nt] * Hdim + mbase) = hb;
    }
}

// ---------------------------------------------------------------------------
// Epilogue: p = W_ph @ h + b_p (10x1024 · 1024xB), then log_softmax rows.
// One thread per batch element; W_ph rows broadcast across the wave.
// ---------------------------------------------------------------------------
__global__ __launch_bounds__(256)
void proj_logsoftmax(const __bf16* __restrict__ hT,   // (B,H) bf16, final h
                     const float* __restrict__ Wph,   // (C,H)
                     const float* __restrict__ bp,    // (C,1)
                     float* __restrict__ out)         // (B,C)
{
    const int b = blockIdx.x * 256 + threadIdx.x;
    float p[Cdim];
    #pragma unroll
    for (int c = 0; c < Cdim; ++c) p[c] = bp[c];
    for (int k = 0; k < Hdim; ++k) {
        const float hv = (float)hT[(size_t)b * Hdim + k];
        #pragma unroll
        for (int c = 0; c < Cdim; ++c) p[c] = fmaf(Wph[c * Hdim + k], hv, p[c]);
    }
    float m = p[0];
    #pragma unroll
    for (int c = 1; c < Cdim; ++c) m = fmaxf(m, p[c]);
    float s = 0.0f;
    #pragma unroll
    for (int c = 0; c < Cdim; ++c) s += __expf(p[c] - m);
    const float lse = m + __logf(s);
    #pragma unroll
    for (int c = 0; c < Cdim; ++c) out[b * Cdim + c] = p[c] - lse;
}

// ---------------------------------------------------------------------------
extern "C" void kernel_launch(void* const* d_in, const int* in_sizes, int n_in,
                              void* d_out, int out_size, void* d_ws, size_t ws_size,
                              hipStream_t stream)
{
    const int*   x   = (const int*)  d_in[0];
    const float* emb = (const float*)d_in[1];
    const float* Wgx = (const float*)d_in[2];
    const float* Wgh = (const float*)d_in[3];
    const float* bg  = (const float*)d_in[4];
    const float* Wix = (const float*)d_in[5];
    const float* Wih = (const float*)d_in[6];
    const float* bi  = (const float*)d_in[7];
    const float* Wfx = (const float*)d_in[8];
    const float* Wfh = (const float*)d_in[9];
    const float* bf  = (const float*)d_in[10];
    const float* Wox = (const float*)d_in[11];
    const float* Woh = (const float*)d_in[12];
    const float* bo  = (const float*)d_in[13];
    const float* Wph = (const float*)d_in[14];
    const float* bp  = (const float*)d_in[15];
    const float* h0  = (const float*)d_in[16];
    const float* c0  = (const float*)d_in[17];

    char* ws = (char*)d_ws;
    __bf16* Wbf   = (__bf16*)(ws);                                 // 8 MB
    __bf16* hT0   = (__bf16*)(ws + 8  * 1024 * 1024);              // 2 MB
    __bf16* hT1   = (__bf16*)(ws + 10 * 1024 * 1024);              // 2 MB
    float*  cT    = (float*) (ws + 12 * 1024 * 1024);              // 4 MB
    float*  xproj = (float*) (ws + 16 * 1024 * 1024);              // 48 KB

    conv_weights<<<(Hdim * Hdim) / 256, 256, 0, stream>>>(Wgh, Wih, Wfh, Woh, Wbf);
    xproj_build <<<(4 * Vdim * Hdim) / 256, 256, 0, stream>>>(
        Wgx, bg, Wix, bi, Wfx, bf, Wox, bo, emb, xproj);
    init_state  <<<(Bsz * Hdim) / 256, 256, 0, stream>>>(h0, c0, hT0, cT);

    __bf16* hbuf[2] = { hT0, hT1 };
    const dim3 grid(Bsz / 64, Hdim / 64);
    for (int t = 0; t < Ssz - 1; ++t) {
        lstm_step<<<grid, 256, 0, stream>>>(
            Wbf, hbuf[t & 1], hbuf[(t + 1) & 1], cT, x, xproj, t);
    }
    proj_logsoftmax<<<Bsz / 256, 256, 0, stream>>>(hbuf[(Ssz - 1) & 1], Wph, bp,
                                                   (float*)d_out);
}
